// TransformerMoE_69587060130092
// MI455X (gfx1250) — compile-verified, matrix operand order checked by reading, not run on hardware
//
#include <hip/hip_runtime.h>
#include <hip/hip_bf16.h>

// ---------------- model constants ----------------
constexpr int Vv  = 32000;
constexpr int Dd  = 1024;
constexpr int Hh  = 16;
constexpr int HDd = 64;
constexpr int Ll  = 3;
constexpr int DFFd= 4096;
constexpr int Ee  = 8;
constexpr int Bb  = 2;
constexpr int Ss  = 1024;
constexpr int Nn  = Bb * Ss;   // 2048 tokens

typedef __attribute__((ext_vector_type(16))) _Float16 v16h;
typedef __attribute__((ext_vector_type(8)))  float    v8f;

// Load a 16x32 f16 WMMA operand fragment from an LDS tile.
// ISA 7.12.2 16-bit A layout: lane L -> row (L&15); lanes 0-15 hold K=0..7,16..23,
// lanes 16-31 hold K=8..15,24..31; each VGPR holds a contiguous (K,K+1) f16 pair
// -> two ds_load_b128 per fragment after compiler coalescing.
// B tiles are staged transposed ([n][k]) so the same loader serves A and B.
__device__ __forceinline__ v16h ld_frag(const _Float16* base, int ldk) {
  const int lane = threadIdx.x & 31;
  const int r    = lane & 15;
  const int kb   = (lane >> 4) << 3;   // 0 or 8
  const _Float16* rp = base + r * ldk;
  union { v16h v; unsigned u[8]; } f;
#pragma unroll
  for (int i = 0; i < 4; ++i)
    f.u[i] = *(const unsigned*)(rp + kb + 2 * i);
#pragma unroll
  for (int i = 0; i < 4; ++i)
    f.u[4 + i] = *(const unsigned*)(rp + 16 + kb + 2 * i);
  return f.v;
}

// ---------------- generic WMMA GEMM ----------------
// C[M,Nc] = epilogue( A[M,Kd](f32,row) x B(f32) + bias ), f16 WMMA, f32 accum.
// 128x128 macro-tile, 256 threads = 8 waves; wave (rg,cg) owns a 32x64 sub-tile
// (2 A-frags x 4 B-frags -> 8 WMMAs per 32-K step, 2 barriers per step).
// flags: bit0 = ReLU, bit1 = B is [Nc,Kd] row-major (transB), bit2 = C += result.
// rowScale: optional per-output-row multiplier with stride rsStride (MoE gates).
// Requires M%128==0, Nc%128==0, Kd%32==0 (true for all call sites here).
constexpr int LDT = 40;   // padded K stride of LDS tiles (even -> dword-aligned pairs)
constexpr int BMt = 128;
constexpr int BNt = 128;

__global__ __launch_bounds__(256)
void gemm_wmma(const float* __restrict__ A, int lda,
               const float* __restrict__ Bw, int ldb,
               const float* __restrict__ bias,
               float* __restrict__ C, int ldc,
               int M, int Nc, int Kd, int flags,
               const float* __restrict__ rowScale, int rsStride)
{
  __shared__ _Float16 As[BMt * LDT];
  __shared__ _Float16 Bs[BNt * LDT];
  const int n0 = blockIdx.x * BNt;
  const int m0 = blockIdx.y * BMt;
  const int t  = threadIdx.x;
  const int w  = t >> 5;
  const int rg = w >> 1;        // row group 0..3  -> rows rg*32 .. rg*32+31
  const int cg = w & 1;         // col group 0..1  -> cols cg*64 .. cg*64+63
  const bool transB = (flags & 2) != 0;

  v8f acc[2][4] = {};

  for (int k0 = 0; k0 < Kd; k0 += 32) {
    // stage A tile 128x32, converting f32 -> f16 (single-pass weight/act reads)
#pragma unroll
    for (int i = 0; i < 16; ++i) {
      int idx = t + i * 256;
      int r = idx >> 5, c = idx & 31;
      As[r * LDT + c] = (_Float16)A[(size_t)(m0 + r) * lda + k0 + c];
    }
    // stage B tile as [n][k]
    if (transB) {
#pragma unroll
      for (int i = 0; i < 16; ++i) {
        int idx = t + i * 256;
        int r = idx >> 5, c = idx & 31;
        Bs[r * LDT + c] = (_Float16)Bw[(size_t)(n0 + r) * ldb + k0 + c];
      }
    } else {
#pragma unroll
      for (int i = 0; i < 16; ++i) {
        int idx = t + i * 256;
        int kk = idx >> 7, nn = idx & 127;
        Bs[nn * LDT + kk] = (_Float16)Bw[(size_t)(k0 + kk) * ldb + n0 + nn];
      }
    }
    // prefetch next K-tile (global_prefetch_b8; weights are streamed once,
    // so L2 warm-up of the next tile is the only useful caching)
    if (k0 + 32 < Kd) {
      __builtin_prefetch(&A[(size_t)(m0 + (t >> 1)) * lda + k0 + 32 + (t & 1) * 16], 0, 1);
      if (transB)
        __builtin_prefetch(&Bw[(size_t)(n0 + (t >> 1)) * ldb + k0 + 32 + (t & 1) * 16], 0, 1);
      else
        __builtin_prefetch(&Bw[(size_t)(k0 + 32 + (t >> 3)) * ldb + n0 + (t & 7) * 16], 0, 1);
    }
    __syncthreads();

    v16h a0 = ld_frag(&As[(rg * 32) * LDT], LDT);
    v16h a1 = ld_frag(&As[(rg * 32 + 16) * LDT], LDT);
#pragma unroll
    for (int j = 0; j < 4; ++j) {
      v16h b = ld_frag(&Bs[(cg * 64 + j * 16) * LDT], LDT);
      acc[0][j] = __builtin_amdgcn_wmma_f32_16x16x32_f16(
          false, a0, false, b, (short)0, acc[0][j], false, false);
      acc[1][j] = __builtin_amdgcn_wmma_f32_16x16x32_f16(
          false, a1, false, b, (short)0, acc[1][j], false, false);
    }
    __syncthreads();
  }

  // epilogue (ISA C layout: lanes 0-15 rows r, lanes 16-31 rows r+8; col = lane&15)
  const int lane = t & 31;
  const int nlo  = lane & 15;
  const int mhi  = (lane >> 4) << 3;
#pragma unroll
  for (int mi = 0; mi < 2; ++mi) {
#pragma unroll
    for (int j = 0; j < 4; ++j) {
#pragma unroll
      for (int r = 0; r < 8; ++r) {
        int gm = m0 + rg * 32 + mi * 16 + mhi + r;
        int gn = n0 + cg * 64 + j * 16 + nlo;
        float val = acc[mi][j][r];
        if (bias) val += bias[gn];
        if (flags & 1) val = fmaxf(val, 0.f);
        if (rowScale) val *= rowScale[(size_t)gm * rsStride];
        size_t o = (size_t)gm * ldc + gn;
        if (flags & 4) C[o] += val; else C[o] = val;
      }
    }
  }
}

// ---------------- embedding gather ----------------
__global__ __launch_bounds__(256)
void embed_k(const int* __restrict__ src, const float* __restrict__ emb,
             float* __restrict__ x)
{
  int n = blockIdx.x, t = threadIdx.x;
  int tok = src[n];
  tok = tok < 0 ? 0 : (tok > Vv - 1 ? Vv - 1 : tok);
#pragma unroll
  for (int i = 0; i < 4; ++i) {
    int d = t + i * 256;
    x[(size_t)n * Dd + d] = emb[(size_t)tok * Dd + d] * 32.0f;  // sqrt(1024)
  }
}

// ---------------- RoPE (in place on q,k) ----------------
__global__ void rope_k(float* __restrict__ q, float* __restrict__ k)
{
  int gid = blockIdx.x * blockDim.x + threadIdx.x;   // over N*H*(HD/2)
  if (gid >= Nn * Hh * (HDd / 2)) return;
  int j = gid & 31;
  int h = (gid >> 5) & (Hh - 1);
  int n = gid >> 9;
  int s = n & (Ss - 1);
  float inv = __powf(10000.f, -(float)(2 * j) * (1.f / (float)HDd));
  float si, c;
  __sincosf((float)s * inv, &si, &c);
  size_t o = (size_t)n * Dd + h * HDd + j;
  float a = q[o], b = q[o + 32];
  q[o]      = a * c - b * si;
  q[o + 32] = b * c + a * si;
  a = k[o]; b = k[o + 32];
  k[o]      = a * c - b * si;
  k[o + 32] = b * c + a * si;
}

// ---------------- flash attention (WMMA) ----------------
// block = 128 threads (4 waves); grid = (S/64, H, B). Each block: 64 q-rows.
// Scores never touch HBM (S^2 tile lives in regs/LDS); online softmax state
// (m,l) kept per-row in registers with 16-lane shfl_xor reductions matching
// the WMMA C layout (lanes 0-15 rows r, lanes 16-31 rows r+8).
__global__ __launch_bounds__(128)
void flash_attn(const float* __restrict__ q, const float* __restrict__ k,
                const float* __restrict__ v, float* __restrict__ o, float scale)
{
  __shared__ _Float16 Qs[64 * 72];
  __shared__ _Float16 Ks[64 * 72];
  __shared__ _Float16 Vs[64 * 72];   // transposed: [hd][sk]
  __shared__ _Float16 Ps[64 * 72];
  const int qt = blockIdx.x, h = blockIdx.y, b = blockIdx.z;
  const int t = threadIdx.x, w = t >> 5, lane = t & 31;
  const int nlo = lane & 15, mhi = (lane >> 4) << 3;

  // stage Q (scale folded in)
#pragma unroll
  for (int i = 0; i < 32; ++i) {
    int idx = t + i * 128;
    int r = idx >> 6, c = idx & 63;
    Qs[r * 72 + c] =
        (_Float16)(q[((size_t)(b * Ss + qt * 64 + r)) * Dd + h * HDd + c] * scale);
  }

  v8f oacc[4] = {};
  float mrow[8], lrow[8];
#pragma unroll
  for (int r = 0; r < 8; ++r) { mrow[r] = -1e30f; lrow[r] = 0.f; }

  for (int kt = 0; kt < Ss / 64; ++kt) {
    // stage K natural ([sk][hd]) and V transposed ([hd][sk])
#pragma unroll
    for (int i = 0; i < 32; ++i) {
      int idx = t + i * 128;
      int r = idx >> 6, c = idx & 63;
      size_t g = ((size_t)(b * Ss + kt * 64 + r)) * Dd + h * HDd + c;
      Ks[r * 72 + c] = (_Float16)k[g];
      Vs[c * 72 + r] = (_Float16)v[g];
    }
    // prefetch next K/V tile
    if (kt + 1 < Ss / 64) {
      size_t gn = ((size_t)(b * Ss + (kt + 1) * 64 + (t >> 1))) * Dd + h * HDd + (t & 1) * 32;
      __builtin_prefetch(&k[gn], 0, 1);
      __builtin_prefetch(&v[gn], 0, 1);
    }
    __syncthreads();

    // S-tile = Q x K^T  (HD=64 -> two k-steps)
    v8f sacc[4] = {};
#pragma unroll
    for (int ks = 0; ks < 64; ks += 32) {
      v16h a = ld_frag(&Qs[(w * 16) * 72 + ks], 72);
#pragma unroll
      for (int j = 0; j < 4; ++j) {
        v16h bf = ld_frag(&Ks[(j * 16) * 72 + ks], 72);
        sacc[j] = __builtin_amdgcn_wmma_f32_16x16x32_f16(
            false, a, false, bf, (short)0, sacc[j], false, false);
      }
    }

    // online softmax: rows owned per C-layout; reduce across 16-lane groups
#pragma unroll
    for (int r = 0; r < 8; ++r) {
      float mx = fmaxf(fmaxf(sacc[0][r], sacc[1][r]),
                       fmaxf(sacc[2][r], sacc[3][r]));
#pragma unroll
      for (int msk = 1; msk < 16; msk <<= 1)
        mx = fmaxf(mx, __shfl_xor(mx, msk, 32));
      float mnew = fmaxf(mrow[r], mx);
      float corr = __expf(mrow[r] - mnew);
      float ls = 0.f;
#pragma unroll
      for (int j = 0; j < 4; ++j) {
        float p = __expf(sacc[j][r] - mnew);
        ls += p;
        Ps[(w * 16 + mhi + r) * 72 + j * 16 + nlo] = (_Float16)p;
      }
#pragma unroll
      for (int msk = 1; msk < 16; msk <<= 1)
        ls += __shfl_xor(ls, msk, 32);
      lrow[r] = lrow[r] * corr + ls;
      mrow[r] = mnew;
#pragma unroll
      for (int j = 0; j < 4; ++j) oacc[j][r] *= corr;
    }
    __syncthreads();

    // O += P x V
#pragma unroll
    for (int ks = 0; ks < 64; ks += 32) {
      v16h a = ld_frag(&Ps[(w * 16) * 72 + ks], 72);
#pragma unroll
      for (int j = 0; j < 4; ++j) {
        v16h bf = ld_frag(&Vs[(j * 16) * 72 + ks], 72);
        oacc[j] = __builtin_amdgcn_wmma_f32_16x16x32_f16(
            false, a, false, bf, (short)0, oacc[j], false, false);
      }
    }
    __syncthreads();
  }

#pragma unroll
  for (int j = 0; j < 4; ++j)
#pragma unroll
    for (int r = 0; r < 8; ++r) {
      int row = qt * 64 + w * 16 + mhi + r;
      o[((size_t)(b * Ss + row)) * Dd + h * HDd + j * 16 + nlo] =
          oacc[j][r] / lrow[r];
    }
}

// ---------------- residual + layernorm ----------------
__global__ __launch_bounds__(256)
void add_ln(const float* __restrict__ x, const float* __restrict__ rz,
            const float* __restrict__ g, const float* __restrict__ bt,
            float* __restrict__ dst)
{
  __shared__ float s1[256], s2[256];
  const int row = blockIdx.x, t = threadIdx.x;
  float vals[4]; float sum = 0.f, sq = 0.f;
#pragma unroll
  for (int i = 0; i < 4; ++i) {
    int d = t + i * 256;
    float vv = x[(size_t)row * Dd + d];
    if (rz) vv += rz[(size_t)row * Dd + d];
    vals[i] = vv; sum += vv; sq += vv * vv;
  }
  s1[t] = sum; s2[t] = sq; __syncthreads();
  for (int st = 128; st > 0; st >>= 1) {
    if (t < st) { s1[t] += s1[t + st]; s2[t] += s2[t + st]; }
    __syncthreads();
  }
  float mu = s1[0] * (1.f / Dd);
  float var = s2[0] * (1.f / Dd) - mu * mu;
  float rs = rsqrtf(var + 1e-5f);
#pragma unroll
  for (int i = 0; i < 4; ++i) {
    int d = t + i * 256;
    dst[(size_t)row * Dd + d] = (vals[i] - mu) * rs * g[d] + bt[d];
  }
}

// ---------------- MoE gating ----------------
__global__ __launch_bounds__(256)
void gate_k(const float* __restrict__ x, const float* __restrict__ gW,
            const float* __restrict__ gb, float* __restrict__ logits)
{
  int n = blockIdx.x;
  int w = threadIdx.x >> 5, lane = threadIdx.x & 31;   // one wave per expert
  float acc = 0.f;
  for (int d = lane; d < Dd; d += 32)
    acc += x[(size_t)n * Dd + d] * gW[(size_t)d * Ee + w];
#pragma unroll
  for (int msk = 1; msk < 32; msk <<= 1)
    acc += __shfl_xor(acc, msk, 32);
  if (lane == 0) logits[n * Ee + w] = acc + gb[w];
}

__global__ void topk_k(const float* __restrict__ logits, float* __restrict__ wfull)
{
  int n = blockIdx.x * blockDim.x + threadIdx.x;
  if (n >= Nn) return;
  float l[Ee];
#pragma unroll
  for (int e = 0; e < Ee; ++e) l[e] = logits[n * Ee + e];
  int i1 = 0;
#pragma unroll
  for (int e = 1; e < Ee; ++e) if (l[e] > l[i1]) i1 = e;
  int i2 = (i1 == 0) ? 1 : 0;
#pragma unroll
  for (int e = 0; e < Ee; ++e) if (e != i1 && l[e] > l[i2]) i2 = e;
  float eb = __expf(l[i2] - l[i1]);
  float inv = 1.f / (1.f + eb);
#pragma unroll
  for (int e = 0; e < Ee; ++e) wfull[n * Ee + e] = 0.f;
  wfull[n * Ee + i1] = inv;
  wfull[n * Ee + i2] = eb * inv;
}

// ---------------- host-side orchestration ----------------
extern "C" void kernel_launch(void* const* d_in, const int* in_sizes, int n_in,
                              void* d_out, int out_size, void* d_ws, size_t ws_size,
                              hipStream_t stream) {
  const int*   src = (const int*)  d_in[0];
  const float* emb = (const float*)d_in[1];
  const float* Wq  = (const float*)d_in[2];  const float* bq = (const float*)d_in[3];
  const float* Wk  = (const float*)d_in[4];  const float* bk = (const float*)d_in[5];
  const float* Wv  = (const float*)d_in[6];  const float* bv = (const float*)d_in[7];
  const float* Wo  = (const float*)d_in[8];  const float* bo = (const float*)d_in[9];
  const float* W1  = (const float*)d_in[10]; const float* b1 = (const float*)d_in[11];
  const float* W2  = (const float*)d_in[12]; const float* b2 = (const float*)d_in[13];
  const float* g1  = (const float*)d_in[14]; const float* be1= (const float*)d_in[15];
  const float* g2  = (const float*)d_in[16]; const float* be2= (const float*)d_in[17];
  const float* gW  = (const float*)d_in[18]; const float* gb = (const float*)d_in[19];
  const float* eW1 = (const float*)d_in[20]; const float* eb1= (const float*)d_in[21];
  const float* eW2 = (const float*)d_in[22]; const float* eb2= (const float*)d_in[23];
  const float* fg  = (const float*)d_in[24]; const float* fb = (const float*)d_in[25];
  const float* oW  = (const float*)d_in[26]; const float* ob = (const float*)d_in[27];
  float* outp = (float*)d_out;
  (void)in_sizes; (void)n_in; (void)out_size; (void)ws_size;

  // workspace layout (floats)
  const size_t NDf = (size_t)Nn * Dd;
  float* xb    = (float*)d_ws;
  float* qb    = xb  + NDf;
  float* kb    = qb  + NDf;
  float* vb    = kb  + NDf;
  float* ao    = vb  + NDf;
  float* t1    = ao  + NDf;
  float* hff   = t1  + NDf;                      // Nn x DFF
  float* moe   = hff + (size_t)Nn * DFFd;
  float* glog  = moe + NDf;
  float* wfull = glog + (size_t)Nn * Ee;

  auto gemm = [&](const float* A, int lda, const float* Bm, int ldb,
                  const float* bias, float* C, int ldc,
                  int M, int Nc, int Kd, int flags,
                  const float* rs, int rss) {
    dim3 grid(Nc / BNt, M / BMt);
    gemm_wmma<<<grid, 256, 0, stream>>>(A, lda, Bm, ldb, bias, C, ldc,
                                        M, Nc, Kd, flags, rs, rss);
  };

  embed_k<<<Nn, 256, 0, stream>>>(src, emb, xb);

  for (int l = 0; l < Ll; ++l) {
    const float* wq = Wq + (size_t)l * Dd * Dd;
    const float* wk = Wk + (size_t)l * Dd * Dd;
    const float* wv = Wv + (size_t)l * Dd * Dd;
    const float* wo = Wo + (size_t)l * Dd * Dd;
    gemm(xb, Dd, wq, Dd, bq + l * Dd, qb, Dd, Nn, Dd, Dd, 0, nullptr, 0);
    gemm(xb, Dd, wk, Dd, bk + l * Dd, kb, Dd, Nn, Dd, Dd, 0, nullptr, 0);
    gemm(xb, Dd, wv, Dd, bv + l * Dd, vb, Dd, Nn, Dd, Dd, 0, nullptr, 0);

    rope_k<<<(Nn * Hh * (HDd / 2)) / 256, 256, 0, stream>>>(qb, kb);

    flash_attn<<<dim3(Ss / 64, Hh, Bb), 128, 0, stream>>>(qb, kb, vb, ao, 0.125f);

    gemm(ao, Dd, wo, Dd, bo + l * Dd, t1, Dd, Nn, Dd, Dd, 0, nullptr, 0);
    add_ln<<<Nn, 256, 0, stream>>>(xb, t1, g1 + l * Dd, be1 + l * Dd, xb);

    gemm(xb, Dd, W1 + (size_t)l * Dd * DFFd, DFFd, b1 + l * DFFd,
         hff, DFFd, Nn, DFFd, Dd, /*relu*/1, nullptr, 0);
    gemm(hff, DFFd, W2 + (size_t)l * DFFd * Dd, Dd, b2 + l * Dd,
         t1, Dd, Nn, Dd, DFFd, 0, nullptr, 0);
    add_ln<<<Nn, 256, 0, stream>>>(xb, t1, g2 + l * Dd, be2 + l * Dd, xb);
  }

  // MoE gating (top-2 of 8)
  gate_k<<<Nn, 256, 0, stream>>>(xb, gW, gb, glog);
  topk_k<<<(Nn + 255) / 256, 256, 0, stream>>>(glog, wfull);

  // dense experts, gate-weighted accumulation fused into GEMM epilogue
  for (int e = 0; e < Ee; ++e) {
    gemm(xb, Dd, eW1 + (size_t)e * Dd * DFFd, DFFd, eb1 + e * DFFd,
         hff, DFFd, Nn, DFFd, Dd, /*relu*/1, nullptr, 0);
    gemm(hff, DFFd, eW2 + (size_t)e * DFFd * Dd, Dd, eb2 + e * Dd,
         moe, Dd, Nn, Dd, DFFd, /*acc*/(e ? 4 : 0), wfull + e, Ee);
  }

  add_ln<<<Nn, 256, 0, stream>>>(moe, nullptr, fg, fb, moe);

  // final vocab projection -> d_out [Nn, Vv]
  gemm(moe, Dd, oW, Vv, ob, outp, Vv, Nn, Vv, Dd, 0, nullptr, 0);
}
